// Location_Sensitive_Stepwise_Monotonic_Attention_32727650796123
// MI455X (gfx1250) — compile-verified, matrix operand order checked by reading, not run on hardware
//
#include <hip/hip_runtime.h>
#include <hip/hip_bf16.h>
#include <cstdint>

typedef __attribute__((ext_vector_type(16))) __bf16 v16bf;
typedef __attribute__((ext_vector_type(8)))  float  v8f;
typedef __attribute__((ext_vector_type(2)))  float  v2f;
typedef __attribute__((ext_vector_type(4)))  unsigned int u32x4;
typedef __attribute__((ext_vector_type(8)))  int    i32x8;
typedef __attribute__((ext_vector_type(4)))  int    i32x4;

#define BB   16
#define EE   512
#define QT   512
#define KT   1024
#define CC   32
#define KW   31
#define PADW 15

#if __has_builtin(__builtin_amdgcn_tensor_load_to_lds) && \
    __has_builtin(__builtin_amdgcn_s_wait_tensorcnt)
#define USE_TDM 1
#else
#define USE_TDM 0
#endif

// ---------------------------------------------------------------------------
// Tensor Data Mover: 2D tile (tile_w elems wide, rows tall, row stride in
// elements) of f32 from global -> LDS. D# layout per CDNA5 ISA §8.3/8.4.
// This toolchain exposes the 6-arg builtin: (g0, g1, g2, g3, g4, cpol).
// ---------------------------------------------------------------------------
#if USE_TDM
__device__ __forceinline__ unsigned ldsOff(const void* p) {
  return (unsigned)(uintptr_t)p;  // LDS aperture: low 32 bits = LDS byte offset
}

__device__ __forceinline__ void tdm_load_2d_f32(unsigned lds_byte_off,
                                                const float* gptr,
                                                unsigned tile_w, unsigned rows,
                                                unsigned long long stride_elems) {
  unsigned long long ga = (unsigned long long)(uintptr_t)gptr;
  u32x4 g0;
  g0[0] = 1u;                                                // count=1 (user D#)
  g0[1] = lds_byte_off;                                      // lds_addr
  g0[2] = (unsigned)(ga & 0xFFFFFFFFu);                      // global_addr[31:0]
  g0[3] = (unsigned)((ga >> 32) & 0x1FFFFFFu) | (2u << 30);  // addr[56:32] | type=2
  // group1 bit layout: [17:16]=data_size(2->4B) [79:48]=tensor_dim0
  // [111:80]=tensor_dim1 [127:112]=tile_dim0 [143:128]=tile_dim1
  // [207:160]=tensor_dim0_stride
  unsigned long long q0 = (2ull << 16) | ((unsigned long long)tile_w << 48);
  unsigned long long q1 = ((unsigned long long)rows << 16) |
                          ((unsigned long long)tile_w << 48);
  unsigned long long q2 = (unsigned long long)rows |
                          ((stride_elems & 0xFFFFFFFFull) << 32);
  unsigned long long q3 = (stride_elems >> 32) & 0xFFFFull;
  i32x8 g1;
  g1[0] = (int)(q0 & 0xFFFFFFFFull); g1[1] = (int)(q0 >> 32);
  g1[2] = (int)(q1 & 0xFFFFFFFFull); g1[3] = (int)(q1 >> 32);
  g1[4] = (int)(q2 & 0xFFFFFFFFull); g1[5] = (int)(q2 >> 32);
  g1[6] = (int)(q3 & 0xFFFFFFFFull); g1[7] = (int)(q3 >> 32);
  i32x4 z4 = {0, 0, 0, 0};
  i32x8 z8 = {0, 0, 0, 0, 0, 0, 0, 0};
  __builtin_amdgcn_tensor_load_to_lds(g0, g1, z4, z4, z8, 0);
}
#endif

// ---------------------------------------------------------------------------
// Converters
// ---------------------------------------------------------------------------
__global__ void cvt_f32_bf16(const float* __restrict__ src, __bf16* __restrict__ dst, int n) {
  int i = blockIdx.x * blockDim.x + threadIdx.x;
  if (i < n) dst[i] = (__bf16)src[i];
}

// src [B][D][T] (fp32) -> dst [B][T][D] (bf16), so K(=D) is contiguous per column
__global__ void transpose_cvt_bf16(const float* __restrict__ src, __bf16* __restrict__ dst,
                                   int d_n, int t_n) {
  int i = blockIdx.x * blockDim.x + threadIdx.x;
  int total = BB * d_n * t_n;
  if (i >= total) return;
  int t = i % t_n;
  int d = (i / t_n) % d_n;
  int b = i / (t_n * d_n);
  dst[((size_t)b * t_n + t) * d_n + d] = (__bf16)src[i];
}

// ---------------------------------------------------------------------------
// Projection GEMM: Y[b][m][n] = sum_d W[m][d] * X[b][n][d]   (bf16 in, f32 out)
// ---------------------------------------------------------------------------
__global__ __launch_bounds__(128)
void gemm_bf16_proj(const __bf16* __restrict__ W,  // [M][K] row-major
                    const __bf16* __restrict__ X,  // [B][N][K] row-major
                    float* __restrict__ Y,         // [B][M][N]
                    int M, int N, int Kdim) {
  const int lane = threadIdx.x & 31;
  const int wave = threadIdx.x >> 5;
  const int b = blockIdx.x;
  const int m0 = blockIdx.y * 16;
  const int n0 = (blockIdx.z * 4 + wave) * 16;
  const int kb = (lane < 16) ? 0 : 8;               // 16-bit A/B fragment K base per lane-half
  const __bf16* Arow = W + (size_t)(m0 + (lane & 15)) * Kdim;
  const __bf16* Brow = X + ((size_t)b * N + (n0 + (lane & 15))) * Kdim;
  v8f acc = {};
  for (int k0 = 0; k0 < Kdim; k0 += 32) {
    __builtin_prefetch(Arow + k0 + 128, 0, 0);
    __builtin_prefetch(Brow + k0 + 128, 0, 0);
    v16bf a, bf;
#pragma unroll
    for (int i = 0; i < 8; ++i) {
      int kk = k0 + kb + ((i >= 4) ? 16 : 0) + 2 * (i & 3);
      a[2 * i]      = Arow[kk];
      a[2 * i + 1]  = Arow[kk + 1];
      bf[2 * i]     = Brow[kk];
      bf[2 * i + 1] = Brow[kk + 1];
    }
    acc = __builtin_amdgcn_wmma_f32_16x16x32_bf16(false, a, false, bf, (short)0, acc,
                                                  false, false);
  }
  float* Yb = Y + (size_t)b * M * N;
  const int n = n0 + (lane & 15);
  const int mbase = m0 + ((lane < 16) ? 0 : 8);
#pragma unroll
  for (int r = 0; r < 8; ++r) Yb[(size_t)(mbase + r) * N + n] = acc[r];
}

// ---------------------------------------------------------------------------
// Scan state init: prev = cum = delta(kt==0)
// ---------------------------------------------------------------------------
__global__ void init_state(float* __restrict__ prev0, float* __restrict__ prev1,
                           float* __restrict__ cum) {
  int i = blockIdx.x * blockDim.x + threadIdx.x;
  if (i < BB * KT) {
    float v = ((i % KT) == 0) ? 1.0f : 0.0f;
    prev0[i] = v;
    prev1[i] = 0.0f;
    cum[i] = v;
  }
}

// ---------------------------------------------------------------------------
// Per-step fused kernel: TDM-stage q/k -> conv0+GELU -> WMMA 1x1 conv ->
// LayerNorm -> tanh score.  grid (B, KT/16), 256 threads (8 waves).
// ---------------------------------------------------------------------------
__global__ __launch_bounds__(256)
void step_scores(const float* __restrict__ qf,       // [B][E][QT]
                 const float* __restrict__ kf,       // [B][E][KT]
                 const __bf16* __restrict__ conv1B,  // [E][C]
                 const float* __restrict__ conv0w,   // [C][2][KW]
                 const float* __restrict__ gamma,
                 const float* __restrict__ beta,
                 const float* __restrict__ scorew,
                 const float* __restrict__ prev_in,  // [B][KT]
                 const float* __restrict__ cum_in,   // [B][KT]
                 float* __restrict__ pbuf,           // [B][KT]
                 int t) {
  __shared__ float  s_cw[CC * 2 * KW];
  __shared__ float  s_prev[48];
  __shared__ float  s_cum[48];
  __shared__ __bf16 s_g[16][CC];       // B-matrix for WMMA: [col j][channel c]
  __shared__ float  s_h[EE * 16];      // h tile [e][j], 32 KB
  __shared__ float  s_q[EE];           // q column (TDM staged)
  __shared__ float  s_k[EE * 16];      // k tile [e][j] (TDM staged), 32 KB
  __shared__ float  s_sum[16], s_sumsq[16], s_score[16], s_mu[16], s_rstd[16];

  const int tid = threadIdx.x;
  const int b = blockIdx.x;
  const int j0 = blockIdx.y * 16;

#if USE_TDM
  // wave 0 kicks off both DMA descriptors; they overlap conv + WMMA below
  if (tid < 32) {
    tdm_load_2d_f32(ldsOff(s_q), qf + (size_t)b * EE * QT + t,
                    /*tile_w=*/1, /*rows=*/EE, /*stride=*/QT);
    tdm_load_2d_f32(ldsOff(s_k), kf + (size_t)b * EE * KT + j0,
                    /*tile_w=*/16, /*rows=*/EE, /*stride=*/KT);
  }
#else
  for (int i = tid; i < EE; i += 256)
    s_q[i] = qf[(size_t)b * EE * QT + (size_t)i * QT + t];
  for (int i = tid; i < EE * 16; i += 256) {
    int e = i >> 4, j = i & 15;
    s_k[i] = kf[(size_t)b * EE * KT + (size_t)e * KT + j0 + j];
  }
#endif

  for (int i = tid; i < CC * 2 * KW; i += 256) s_cw[i] = conv0w[i];
  if (tid < 46) {
    int jj = j0 - PADW + tid;
    bool ok = (jj >= 0) && (jj < KT);
    s_prev[tid] = ok ? prev_in[b * KT + jj] : 0.0f;
    s_cum[tid]  = ok ? cum_in[b * KT + jj] : 0.0f;
  }
  if (tid < 16) { s_sum[tid] = 0.f; s_sumsq[tid] = 0.f; s_score[tid] = 0.f; }
  __syncthreads();

  // location conv (C=32 out ch, 2 in ch, 31 taps) + tanh-GELU -> bf16 B fragment
  for (int idx = tid; idx < CC * 16; idx += 256) {
    int c = idx >> 4, j = idx & 15;
    const float* w0 = &s_cw[c * 2 * KW];
    float acc = 0.0f;
    for (int kk = 0; kk < KW; ++kk)
      acc += w0[kk] * s_prev[j + kk] + w0[KW + kk] * s_cum[j + kk];
    float g = 0.5f * acc *
              (1.0f + tanhf(0.7978845608028654f * (acc + 0.044715f * acc * acc * acc)));
    s_g[j][c] = (__bf16)g;
  }
  __syncthreads();

  // h = conv1_w (E x 32) * g (32 x 16): one bf16 WMMA (K=32) per 16x16 e-tile
  const int lane = tid & 31;
  const int wave = tid >> 5;
  const int kb = (lane < 16) ? 0 : 8;
  const int jcol = lane & 15;
  float psum = 0.0f, psumsq = 0.0f;
#pragma unroll
  for (int it = 0; it < 4; ++it) {
    int etile = wave * 4 + it;                       // 8 waves x 4 = 32 e-tiles
    const __bf16* Arow = conv1B + (size_t)(etile * 16 + (lane & 15)) * CC;
    const __bf16* Bcol = &s_g[jcol][0];
    v16bf a, bf;
#pragma unroll
    for (int i = 0; i < 8; ++i) {
      int kk = kb + ((i >= 4) ? 16 : 0) + 2 * (i & 3);
      a[2 * i]      = Arow[kk];
      a[2 * i + 1]  = Arow[kk + 1];
      bf[2 * i]     = Bcol[kk];
      bf[2 * i + 1] = Bcol[kk + 1];
    }
    v8f acc = {};
    acc = __builtin_amdgcn_wmma_f32_16x16x32_bf16(false, a, false, bf, (short)0, acc,
                                                  false, false);
    int mbase = etile * 16 + ((lane < 16) ? 0 : 8);
#pragma unroll
    for (int r = 0; r < 8; ++r) {
      float hv = acc[r];
      s_h[(mbase + r) * 16 + jcol] = hv;
      psum += hv;
      psumsq += hv * hv;
    }
  }
  atomicAdd(&s_sum[jcol], psum);      // ds_add_f32 reduction for LN stats
  atomicAdd(&s_sumsq[jcol], psumsq);
  __syncthreads();

  if (tid < 16) {
    float mu = s_sum[tid] * (1.0f / EE);
    float var = s_sumsq[tid] * (1.0f / EE) - mu * mu;
    s_mu[tid] = mu;
    s_rstd[tid] = rsqrtf(var + 1e-5f);
  }
#if USE_TDM
  if (tid < 32) __builtin_amdgcn_s_wait_tensorcnt(0);  // q/k tiles landed in LDS
#endif
  __syncthreads();

  // normalize + tanh(q_t + k + h) + score-weight dot, reduced over E
  {
    int j = tid & 15, eg = tid >> 4;
    float mu = s_mu[j], rstd = s_rstd[j];
    float acc = 0.0f;
    for (int e = eg * 32; e < eg * 32 + 32; ++e) {
      float hn = (s_h[e * 16 + j] - mu) * rstd * gamma[e] + beta[e];
      float sv = tanhf(s_q[e] + s_k[e * 16 + j] + hn);
      acc += scorew[e] * sv;
    }
    atomicAdd(&s_score[j], acc);
  }
  __syncthreads();

  if (tid < 16) {
    int jj = j0 + tid;
    pbuf[b * KT + jj] = 1.0f / (1.0f + __expf(-s_score[tid]));
  }
}

// ---------------------------------------------------------------------------
// Per-step pointwise recurrence: new = prev*p + shift(prev*(1-p))
// ---------------------------------------------------------------------------
__global__ void step_update(const float* __restrict__ pbuf,
                            const float* __restrict__ prev_in,
                            float* __restrict__ prev_out,
                            float* __restrict__ cum,
                            float* __restrict__ aligns,  // [B][KT][QT]
                            int t) {
  int i = blockIdx.x * blockDim.x + threadIdx.x;
  if (i >= BB * KT) return;
  int b = i / KT, j = i % KT;
  float nm = prev_in[i] * pbuf[i];
  if (j > 0) nm += prev_in[i - 1] * (1.0f - pbuf[i - 1]);
  prev_out[i] = nm;
  cum[i] += nm;
  aligns[((size_t)b * KT + j) * QT + t] = nm;
}

// ---------------------------------------------------------------------------
// contexts[b][e][t] = sum_k v[b][e][k] * align[b][k][t]  — full-fp32 WMMA
// ---------------------------------------------------------------------------
__global__ __launch_bounds__(128)
void contexts_gemm(const float* __restrict__ V,   // [B][E][KT]
                   const float* __restrict__ Al,  // [B][KT][QT]
                   float* __restrict__ Ctx) {     // [B][E][QT]
  const int lane = threadIdx.x & 31;
  const int wave = threadIdx.x >> 5;
  const int b = blockIdx.x;
  const int m0 = blockIdx.y * 16;
  const int n0 = (blockIdx.z * 4 + wave) * 16;
  const int koff = (lane < 16) ? 0 : 2;            // f32 A/B: lane-half selects K/K+2
  const int n = n0 + (lane & 15);
  const float* Arow = V + ((size_t)b * EE + (m0 + (lane & 15))) * KT + koff;
  const float* Bcol = Al + (size_t)b * KT * QT + (size_t)koff * QT + n;
  v8f acc = {};
  for (int k0 = 0; k0 < KT; k0 += 4) {
    __builtin_prefetch(Arow + k0 + 64, 0, 0);
    v2f a = *(const v2f*)(Arow + k0);
    v2f bf;
    bf[0] = Bcol[(size_t)k0 * QT];
    bf[1] = Bcol[(size_t)(k0 + 1) * QT];
    acc = __builtin_amdgcn_wmma_f32_16x16x4_f32(false, a, false, bf, (short)0, acc,
                                                false, false);
  }
  float* Cb = Ctx + (size_t)b * EE * QT;
  const int mbase = m0 + ((lane < 16) ? 0 : 8);
#pragma unroll
  for (int r = 0; r < 8; ++r) Cb[(size_t)(mbase + r) * QT + n] = acc[r];
}

// ---------------------------------------------------------------------------
extern "C" void kernel_launch(void* const* d_in, const int* in_sizes, int n_in,
                              void* d_out, int out_size, void* d_ws, size_t ws_size,
                              hipStream_t stream) {
  const float* queries = (const float*)d_in[0];  // [B][E][QT]
  const float* values  = (const float*)d_in[1];  // [B][E][KT]
  const float* Wq      = (const float*)d_in[2];  // [E][E]
  const float* Wk      = (const float*)d_in[3];
  const float* Wv      = (const float*)d_in[4];
  const float* conv0w  = (const float*)d_in[5];  // [C][2][KW]
  const float* conv1w  = (const float*)d_in[6];  // [E][C]
  const float* gamma   = (const float*)d_in[7];
  const float* beta    = (const float*)d_in[8];
  const float* scorew  = (const float*)d_in[9];

  float* out = (float*)d_out;
  const size_t CTXN = (size_t)BB * EE * QT;      // contexts elements
  float* ctx = out;
  float* aligns = out + CTXN;                    // [B][KT][QT]

  // workspace carve-out
  char* ws = (char*)d_ws;
  size_t off = 0;
  auto take = [&](size_t bytes) -> char* {
    off = (off + 255) & ~(size_t)255;
    char* p = ws + off;
    off += bytes;
    return p;
  };
  float*  qf    = (float*)take((size_t)BB * EE * QT * 4);
  float*  kf    = (float*)take((size_t)BB * EE * KT * 4);
  float*  vf    = (float*)take((size_t)BB * EE * KT * 4);
  __bf16* WqB   = (__bf16*)take((size_t)EE * EE * 2);
  __bf16* WkB   = (__bf16*)take((size_t)EE * EE * 2);
  __bf16* WvB   = (__bf16*)take((size_t)EE * EE * 2);
  __bf16* c1B   = (__bf16*)take((size_t)EE * CC * 2);
  __bf16* qTB   = (__bf16*)take((size_t)BB * QT * EE * 2);
  __bf16* vTB   = (__bf16*)take((size_t)BB * KT * EE * 2);
  float*  prev0 = (float*)take((size_t)BB * KT * 4);
  float*  prev1 = (float*)take((size_t)BB * KT * 4);
  float*  cum   = (float*)take((size_t)BB * KT * 4);
  float*  pbuf  = (float*)take((size_t)BB * KT * 4);
  (void)ws_size; (void)in_sizes; (void)n_in; (void)out_size;

  // 1) converts
  {
    int n = EE * EE;
    cvt_f32_bf16<<<(n + 255) / 256, 256, 0, stream>>>(Wq, WqB, n);
    cvt_f32_bf16<<<(n + 255) / 256, 256, 0, stream>>>(Wk, WkB, n);
    cvt_f32_bf16<<<(n + 255) / 256, 256, 0, stream>>>(Wv, WvB, n);
    int nc = EE * CC;
    cvt_f32_bf16<<<(nc + 255) / 256, 256, 0, stream>>>(conv1w, c1B, nc);
    int nq = BB * EE * QT;
    transpose_cvt_bf16<<<(nq + 255) / 256, 256, 0, stream>>>(queries, qTB, EE, QT);
    int nv = BB * EE * KT;
    transpose_cvt_bf16<<<(nv + 255) / 256, 256, 0, stream>>>(values, vTB, EE, KT);
  }

  // 2) projections q = Wq*queries, k = Wk*values, v = Wv*values  (bf16 WMMA)
  gemm_bf16_proj<<<dim3(BB, EE / 16, QT / 64), 128, 0, stream>>>(WqB, qTB, qf, EE, QT, EE);
  gemm_bf16_proj<<<dim3(BB, EE / 16, KT / 64), 128, 0, stream>>>(WkB, vTB, kf, EE, KT, EE);
  gemm_bf16_proj<<<dim3(BB, EE / 16, KT / 64), 128, 0, stream>>>(WvB, vTB, vf, EE, KT, EE);

  // 3) scan state init
  init_state<<<(BB * KT + 255) / 256, 256, 0, stream>>>(prev0, prev1, cum);

  // 4) sequential monotonic-attention scan, 2 launches per step
  for (int t = 0; t < QT; ++t) {
    float* pin = (t & 1) ? prev1 : prev0;
    float* pout = (t & 1) ? prev0 : prev1;
    step_scores<<<dim3(BB, KT / 16), 256, 0, stream>>>(qf, kf, c1B, conv0w, gamma, beta,
                                                       scorew, pin, cum, pbuf, t);
    step_update<<<(BB * KT + 255) / 256, 256, 0, stream>>>(pbuf, pin, pout, cum, aligns, t);
  }

  // 5) contexts = v @ alignments  (fp32 WMMA for output precision)
  contexts_gemm<<<dim3(BB, EE / 16, QT / 64), 128, 0, stream>>>(vf, aligns, ctx);
}
